// NonLocalBlock_72035191489152
// MI455X (gfx1250) — matrix operand order repaired and checked by reading, not compiled, not run on hardware
//
#include <hip/hip_runtime.h>
#include <hip/hip_bf16.h>
#include <math.h>

typedef __attribute__((ext_vector_type(16))) __bf16 v16bf;
typedef __attribute__((ext_vector_type(8)))  float  v8f;

#define TILE 128
#define KT   32
#define LDK  40   // padded bf16 elems per LDS row: 80B row keeps 16B-aligned b128 loads

// C[M x N] (f32, row stride cs_) = op(A) * op(B), bf16 WMMA, f32 accumulate.
// TA==0: A stored (M,K) row-major (row stride as_).  TA==1: A stored (K,M), transposed while staging.
// TB==0: B stored (K,N) (row stride bs_), transposed while staging.  TB==1: B stored (N,K).
// blockIdx.z = batch with element strides ab_/bb_/cb_. Optional per-row bias.
// Software-pipelined: ping-pong LDS buffers, global loads for tile k+1 issued before tile k's WMMAs.
template<int TA, int TB>
__global__ __launch_bounds__(256) void gemm_bf16_wmma(
    const float* __restrict__ A, const float* __restrict__ B,
    float* __restrict__ C, const float* __restrict__ bias,
    int M, int N, int K,
    long as_, long bs_, long cs_,
    long ab_, long bb_, long cb_)
{
  (void)M; (void)N;
  __shared__ __bf16 Als[2][TILE * LDK];
  __shared__ __bf16 Bls[2][TILE * LDK];

  const int tid  = threadIdx.x;
  const int lane = tid & 31;
  const int wave = tid >> 5;
  const int wm   = wave >> 1;   // 0..3 : 32-row band
  const int wn   = wave & 1;    // 0..1 : 64-col band

  const int m0 = blockIdx.y * TILE;
  const int n0 = blockIdx.x * TILE;

  const float* Ab = A + (long)blockIdx.z * ab_;
  const float* Bb = B + (long)blockIdx.z * bb_;
  float*       Cb = C + (long)blockIdx.z * cb_;

  // ---- staging helpers: load-phase (global -> regs), store-phase (regs -> LDS bf16) ----
  auto loadA = [&](int k0, float4* r) {
    #pragma unroll
    for (int it = 0; it < 4; ++it) {
      const int li = tid + it * 256;
      const float* src;
      if (TA == 0) {                       // 128 rows x 8 float4 of K
        const int rr = li >> 3, kc = (li & 7) << 2;
        src = Ab + (long)(m0 + rr) * as_ + (k0 + kc);
      } else {                             // 32 k-rows x 32 float4 of M
        const int kr = li >> 5, mc = (li & 31) << 2;
        src = Ab + (long)(k0 + kr) * as_ + (m0 + mc);
      }
      if (it == 0 && k0 + KT < K)
        __builtin_prefetch(TA == 0 ? src + KT : src + (long)KT * as_, 0, 1);
      r[it] = *(const float4*)src;
    }
  };
  auto storeA = [&](__bf16* L, const float4* r) {
    #pragma unroll
    for (int it = 0; it < 4; ++it) {
      const int li = tid + it * 256;
      const float4 v = r[it];
      if (TA == 0) {
        const int rr = li >> 3, kc = (li & 7) << 2;
        __bf16* d = &L[rr * LDK + kc];
        d[0] = (__bf16)v.x; d[1] = (__bf16)v.y; d[2] = (__bf16)v.z; d[3] = (__bf16)v.w;
      } else {
        const int kr = li >> 5, mc = (li & 31) << 2;
        L[(mc + 0) * LDK + kr] = (__bf16)v.x;
        L[(mc + 1) * LDK + kr] = (__bf16)v.y;
        L[(mc + 2) * LDK + kr] = (__bf16)v.z;
        L[(mc + 3) * LDK + kr] = (__bf16)v.w;
      }
    }
  };
  auto loadB = [&](int k0, float4* r) {
    #pragma unroll
    for (int it = 0; it < 4; ++it) {
      const int li = tid + it * 256;
      const float* src;
      if (TB == 1) {
        const int rr = li >> 3, kc = (li & 7) << 2;
        src = Bb + (long)(n0 + rr) * bs_ + (k0 + kc);
      } else {
        const int kr = li >> 5, nc = (li & 31) << 2;
        src = Bb + (long)(k0 + kr) * bs_ + (n0 + nc);
      }
      if (it == 0 && k0 + KT < K)
        __builtin_prefetch(TB == 1 ? src + KT : src + (long)KT * bs_, 0, 1);
      r[it] = *(const float4*)src;
    }
  };
  auto storeB = [&](__bf16* L, const float4* r) {
    #pragma unroll
    for (int it = 0; it < 4; ++it) {
      const int li = tid + it * 256;
      const float4 v = r[it];
      if (TB == 1) {
        const int rr = li >> 3, kc = (li & 7) << 2;
        __bf16* d = &L[rr * LDK + kc];
        d[0] = (__bf16)v.x; d[1] = (__bf16)v.y; d[2] = (__bf16)v.z; d[3] = (__bf16)v.w;
      } else {
        const int kr = li >> 5, nc = (li & 31) << 2;
        L[(nc + 0) * LDK + kr] = (__bf16)v.x;
        L[(nc + 1) * LDK + kr] = (__bf16)v.y;
        L[(nc + 2) * LDK + kr] = (__bf16)v.z;
        L[(nc + 3) * LDK + kr] = (__bf16)v.w;
      }
    }
  };

  v8f acc[2][4] = {};
  float4 ra[4], rb[4];

  // LDS fragment base offsets (constant across K loop)
  // A 16x32 bf16 frag: lanes 0-15 = K{0..7,16..23}, lanes 16-31 = K{8..15,24..31}, row = lane%16
  const int kh = (lane >> 4) << 3;   // 0 or 8
  // B 32x16 bf16 frag (column-major staged): lanes 0-15 = K{0..7,16..23}, lanes 16-31 = K{8..15,24..31}, col = lane%16
  const int ks = (lane >> 4) << 3;   // 0 or 8

  const int nk = K / KT;

  // prologue: stage tile 0 into buffer 0
  loadA(0, ra);
  loadB(0, rb);
  storeA(Als[0], ra);
  storeB(Bls[0], rb);
  __syncthreads();

  for (int k = 0; k < nk; ++k) {
    const int cur = k & 1;
    const int nxt = cur ^ 1;
    const bool more = (k + 1) < nk;

    // issue next tile's global loads before this tile's math (latency overlap)
    if (more) {
      loadA((k + 1) * KT, ra);
      loadB((k + 1) * KT, rb);
    }

    // ---- fragments from current buffer + 8 WMMAs ----
    v16bf af[2], bfv[4];
    #pragma unroll
    for (int sm = 0; sm < 2; ++sm) {
      const int row = wm * 32 + sm * 16 + (lane & 15);
      const __bf16* p = &Als[cur][row * LDK + kh];
      __builtin_memcpy(&af[sm], p, 16);
      __builtin_memcpy((char*)&af[sm] + 16, p + 16, 16);
    }
    #pragma unroll
    for (int sn = 0; sn < 4; ++sn) {
      const int col = wn * 64 + sn * 16 + (lane & 15);
      const __bf16* p = &Bls[cur][col * LDK + ks];
      __builtin_memcpy(&bfv[sn], p, 16);
      __builtin_memcpy((char*)&bfv[sn] + 16, p + 16, 16);
    }

    #pragma unroll
    for (int sm = 0; sm < 2; ++sm)
      #pragma unroll
      for (int sn = 0; sn < 4; ++sn)
        acc[sm][sn] = __builtin_amdgcn_wmma_f32_16x16x32_bf16(
            false, af[sm], false, bfv[sn], (short)0, acc[sm][sn], false, false);

    // ---- convert + store next tile into the other buffer ----
    if (more) {
      storeA(Als[nxt], ra);
      storeB(Bls[nxt], rb);
    }
    __syncthreads();
  }

  // ---- write out (D: VGPR r, lanes<16 -> M=r, lanes>=16 -> M=8+r) ----
  const int rb2 = (lane >> 4) ? 8 : 0;
  #pragma unroll
  for (int sm = 0; sm < 2; ++sm) {
    #pragma unroll
    for (int sn = 0; sn < 4; ++sn) {
      const int col = n0 + wn * 64 + sn * 16 + (lane & 15);
      #pragma unroll
      for (int r = 0; r < 8; ++r) {
        const int row = m0 + wm * 32 + sm * 16 + rb2 + r;
        float o = acc[sm][sn][r];
        if (bias) o += bias[row];
        Cb[(long)row * cs_ + col] = o;
      }
    }
  }
}

__global__ __launch_bounds__(256) void softmax_rows(float* __restrict__ f, int ncols) {
  const int tid = threadIdx.x;
  float* p = f + (long)blockIdx.x * ncols;
  __shared__ float red[256];

  float mx = -INFINITY;
  for (int c = tid; c < ncols; c += 256) mx = fmaxf(mx, p[c]);
  red[tid] = mx; __syncthreads();
  for (int s = 128; s > 0; s >>= 1) { if (tid < s) red[tid] = fmaxf(red[tid], red[tid + s]); __syncthreads(); }
  mx = red[0]; __syncthreads();

  float sum = 0.0f;
  for (int c = tid; c < ncols; c += 256) { float e = __expf(p[c] - mx); p[c] = e; sum += e; }
  red[tid] = sum; __syncthreads();
  for (int s = 128; s > 0; s >>= 1) { if (tid < s) red[tid] += red[tid + s]; __syncthreads(); }
  const float inv = 1.0f / red[0];
  for (int c = tid; c < ncols; c += 256) p[c] *= inv;
}

__global__ __launch_bounds__(256) void bn_stats(const float* __restrict__ wy,
                                                float* __restrict__ mean, float* __restrict__ rstd,
                                                int Bn, int Cc, long N) {
  const int tid = threadIdx.x;
  const int c = blockIdx.x;
  __shared__ float s1[256], s2[256];
  float s = 0.0f, ss = 0.0f;
  for (int b = 0; b < Bn; ++b) {
    const float* p = wy + ((long)b * Cc + c) * N;
    for (long i = tid; i < N; i += 256) { float v = p[i]; s += v; ss += v * v; }
  }
  s1[tid] = s; s2[tid] = ss; __syncthreads();
  for (int st = 128; st > 0; st >>= 1) {
    if (tid < st) { s1[tid] += s1[tid + st]; s2[tid] += s2[tid + st]; }
    __syncthreads();
  }
  if (tid == 0) {
    const float cnt = (float)(Bn * N);
    const float m = s1[0] / cnt;
    const float var = s2[0] / cnt - m * m;
    mean[c] = m;
    rstd[c] = rsqrtf(var + 1e-5f);
  }
}

__global__ __launch_bounds__(256) void bn_apply_residual(
    const float* __restrict__ wy, const float* __restrict__ x,
    const float* __restrict__ mean, const float* __restrict__ rstd,
    const float* __restrict__ gamma, const float* __restrict__ beta,
    float* __restrict__ out, int Cc, long N, long total) {
  const long i = (long)blockIdx.x * 256 + threadIdx.x;
  if (i >= total) return;
  const int c = (int)((i / N) % Cc);
  out[i] = (wy[i] - mean[c]) * rstd[c] * gamma[c] + beta[c] + x[i];
}

extern "C" void kernel_launch(void* const* d_in, const int* in_sizes, int n_in,
                              void* d_out, int out_size, void* d_ws, size_t ws_size,
                              hipStream_t stream) {
  (void)in_sizes; (void)n_in; (void)out_size; (void)ws_size;
  const float* x       = (const float*)d_in[0];
  const float* g_w     = (const float*)d_in[1];
  const float* g_b     = (const float*)d_in[2];
  const float* theta_w = (const float*)d_in[3];
  const float* theta_b = (const float*)d_in[4];
  const float* phi_w   = (const float*)d_in[5];
  const float* phi_b   = (const float*)d_in[6];
  const float* w_w     = (const float*)d_in[7];
  const float* w_b     = (const float*)d_in[8];
  const float* gamma   = (const float*)d_in[9];
  const float* beta    = (const float*)d_in[10];
  float* out = (float*)d_out;

  const int  Bn = 4, Cc = 1024, CI = 512;
  const long N = 2048;

  char* ws = (char*)d_ws;
  float* thetaP = (float*)ws; ws += (size_t)Bn * CI * N * 4;   // (b, ci, n)
  float* phiP   = (float*)ws; ws += (size_t)Bn * CI * N * 4;   // (b, ci, n)
  float* gP     = (float*)ws; ws += (size_t)Bn * CI * N * 4;   // (b, ci, n)
  float* fbuf   = (float*)ws; ws += (size_t)Bn * N  * N * 4;   // (b, n, m) -> attn in place
  float* ybuf   = (float*)ws; ws += (size_t)Bn * CI * N * 4;   // (b, ci, n)
  float* wybuf  = (float*)ws; ws += (size_t)Bn * Cc * N * 4;   // (b, c, n)
  float* meanb  = (float*)ws; ws += 1024 * 4;
  float* rstdb  = (float*)ws;

  dim3 blk(256);

  // ---- projections: P[o][n] = W[o][:] . x[:][n]  (A row-major MxK, B is (K,N)) ----
  dim3 gproj((unsigned)(N / TILE), CI / TILE, Bn);
  gemm_bf16_wmma<0, 0><<<gproj, blk, 0, stream>>>(theta_w, x, thetaP, theta_b,
      CI, (int)N, Cc, (long)Cc, N, N, 0L, (long)Cc * N, (long)CI * N);
  gemm_bf16_wmma<0, 0><<<gproj, blk, 0, stream>>>(phi_w, x, phiP, phi_b,
      CI, (int)N, Cc, (long)Cc, N, N, 0L, (long)Cc * N, (long)CI * N);
  gemm_bf16_wmma<0, 0><<<gproj, blk, 0, stream>>>(g_w, x, gP, g_b,
      CI, (int)N, Cc, (long)Cc, N, N, 0L, (long)Cc * N, (long)CI * N);

  // ---- f[n][m] = sum_i theta[i][n] * phi[i][m]   (A stored (K,M) -> TA=1) ----
  dim3 gf((unsigned)(N / TILE), (unsigned)(N / TILE), Bn);
  gemm_bf16_wmma<1, 0><<<gf, blk, 0, stream>>>(thetaP, phiP, fbuf, nullptr,
      (int)N, (int)N, CI, N, N, N, (long)CI * N, (long)CI * N, (long)N * N);

  // ---- softmax over m, in place ----
  softmax_rows<<<dim3((unsigned)(Bn * N)), blk, 0, stream>>>(fbuf, (int)N);

  // ---- y[o][n] = sum_m g[o][m] * attn[n][m]   (B stored (N,K) -> TB=1) ----
  dim3 gy((unsigned)(N / TILE), CI / TILE, Bn);
  gemm_bf16_wmma<0, 1><<<gy, blk, 0, stream>>>(gP, fbuf, ybuf, nullptr,
      CI, (int)N, (int)N, N, N, N, (long)CI * N, (long)N * N, (long)CI * N);

  // ---- w_y[c][n] = sum_o w_w[c][o] * y[o][n] + w_b[c] ----
  dim3 gw((unsigned)(N / TILE), Cc / TILE, Bn);
  gemm_bf16_wmma<0, 0><<<gw, blk, 0, stream>>>(w_w, ybuf, wybuf, w_b,
      Cc, (int)N, CI, (long)CI, N, N, 0L, (long)CI * N, (long)Cc * N);

  // ---- BatchNorm (training stats over batch+spatial) + residual ----
  bn_stats<<<dim3(Cc), blk, 0, stream>>>(wybuf, meanb, rstdb, Bn, Cc, N);
  const long total = (long)Bn * Cc * N;
  bn_apply_residual<<<dim3((unsigned)(total / 256)), blk, 0, stream>>>(
      wybuf, x, meanb, rstdb, gamma, beta, out, Cc, N, total);
}
